// CyGNetModel_20444044329681
// MI455X (gfx1250) — compile-verified, matrix operand order checked by reading, not run on hardware
//
#include <hip/hip_runtime.h>
#include <hip/hip_bf16.h>

#define NUM_ENTS 1000
#define NUM_RELS 100
#define HIDDEN   200
#define NQ       50000
#define NHIST    1000000
#define K_REAL   600
#define K_PAD    608          // 19 * 32
#define H_PAD    208          // 13 * 16
#define KSTEPS   19
#define NWAVES   13
#define TPB      (NWAVES * 32)   // 416 threads

typedef _Float16 v16h __attribute__((ext_vector_type(16)));
typedef float    v8f  __attribute__((ext_vector_type(8)));

union Frag { v16h h; uint4 u[2]; };

// ---------------- prep kernels ----------------

__global__ void zero_u32(unsigned* __restrict__ p, int n) {
  int i = blockIdx.x * blockDim.x + threadIdx.x;
  if (i < n) p[i] = 0u;
}

__global__ void scatter_pairs(const int* __restrict__ hs, const int* __restrict__ hr,
                              unsigned* __restrict__ flag, int n) {
  int i = blockIdx.x * blockDim.x + threadIdx.x;
  if (i < n) flag[hs[i] * NUM_RELS + hr[i]] = 1u;   // idempotent store, no atomic needed
}

// w: [K_REAL, HIDDEN] row-major f32  ->  out: [H_PAD][K_PAD] f16, transposed + zero-padded
__global__ void pack_wt(const float* __restrict__ w, _Float16* __restrict__ out) {
  int i = blockIdx.x * blockDim.x + threadIdx.x;
  if (i >= H_PAD * K_PAD) return;
  int n = i / K_PAD, k = i - n * K_PAD;
  float v = (n < HIDDEN && k < K_REAL) ? w[k * HIDDEN + n] : 0.0f;
  out[i] = (_Float16)v;
}

__global__ void cvt_f16k(const float* __restrict__ in, _Float16* __restrict__ out, int n) {
  int i = blockIdx.x * blockDim.x + threadIdx.x;
  if (i < n) out[i] = (_Float16)in[i];
}

// ---------------- main fused kernel ----------------

__device__ __forceinline__ float wmaxred(float v) {
#pragma unroll
  for (int o = 16; o > 0; o >>= 1) v = fmaxf(v, __shfl_xor(v, o, 32));
  return v;
}
__device__ __forceinline__ float wsumred(float v) {
#pragma unroll
  for (int o = 16; o > 0; o >>= 1) v += __shfl_xor(v, o, 32);
  return v;
}

__global__ __launch_bounds__(TPB)
void cygnet_main(const _Float16* __restrict__ entH, const _Float16* __restrict__ relH,
                 const _Float16* __restrict__ timH,
                 const _Float16* __restrict__ gWt, const _Float16* __restrict__ cWt,
                 const float* __restrict__ g_b, const float* __restrict__ c_b,
                 const unsigned* __restrict__ pairFlag,
                 const int* __restrict__ sub, const int* __restrict__ rel,
                 float* __restrict__ out) {
  __shared__ _Float16 sA[16 * K_PAD];   // 19456 B : x tile (f16)
  __shared__ float    sG[16 * H_PAD];   // 13312 B : g logits
  __shared__ float    sC[16 * H_PAD];   // 13312 B : c logits
  __shared__ float    sMask[16];

  const int t  = threadIdx.x;
  const int mt = blockIdx.x;            // tile of 16 query rows (50000 = 3125*16 exact)

  // ---- phase 1: gather + convert x tile into LDS ----
  for (int i = t; i < 16 * K_PAD; i += TPB) {
    int r = i / K_PAD, k = i - r * K_PAD;
    int rg = mt * 16 + r;
    _Float16 v;
    if (k < HIDDEN)            v = entH[sub[rg] * HIDDEN + k];
    else if (k < 2 * HIDDEN)   v = relH[rel[rg] * HIDDEN + (k - HIDDEN)];
    else if (k < K_REAL)       v = timH[k - 2 * HIDDEN];
    else                       v = (_Float16)0.0f;
    sA[i] = v;
  }
  if (t < 16) {
    int rg = mt * 16 + t;
    sMask[t] = pairFlag[sub[rg] * NUM_RELS + rel[rg]] ? 1.0f : 0.0f;
  }
  __syncthreads();

  // ---- phase 2: dual WMMA GEMM (g and c share the A fragment) ----
  const int wid   = t >> 5;
  const int lane  = t & 31;
  const int m     = lane & 15;          // A row within tile
  const int kbi   = lane >> 4;          // K sub-chunk select (0 -> K+0, 1 -> K+8 halves)
  const int n16   = lane & 15;          // B column within wave tile
  const int cglob = wid * 16 + n16;     // global output column (0..207)

  const uint4* sAq = reinterpret_cast<const uint4*>(sA);
  const uint4* gq  = reinterpret_cast<const uint4*>(gWt + (size_t)cglob * K_PAD);
  const uint4* cq  = reinterpret_cast<const uint4*>(cWt + (size_t)cglob * K_PAD);

  v8f accG = {};
  v8f accC = {};
#pragma unroll
  for (int ks = 0; ks < KSTEPS; ++ks) {
    Frag a, bg, bc;
    const int qi = 4 * ks + kbi;        // uint4 index within a 608-half row (76 uint4)
    a.u[0]  = sAq[m * 76 + qi];
    a.u[1]  = sAq[m * 76 + qi + 2];
    bg.u[0] = gq[qi];
    bg.u[1] = gq[qi + 2];
    bc.u[0] = cq[qi];
    bc.u[1] = cq[qi + 2];
    accG = __builtin_amdgcn_wmma_f32_16x16x32_f16(false, a.h, false, bg.h,
                                                  (short)0, accG, false, false);
    accC = __builtin_amdgcn_wmma_f32_16x16x32_f16(false, a.h, false, bc.h,
                                                  (short)0, accC, false, false);
  }

  // ---- phase 3: accumulators + bias -> LDS logits ----
  float bgv = 0.0f, bcv = 0.0f;
  if (cglob < HIDDEN) { bgv = g_b[cglob]; bcv = c_b[cglob]; }
  const int rbase = (lane < 16) ? 0 : 8;
#pragma unroll
  for (int r = 0; r < 8; ++r) {
    sG[(rbase + r) * H_PAD + cglob] = accG[r] + bgv;
    sC[(rbase + r) * H_PAD + cglob] = accC[r] + bcv;
  }
  __syncthreads();

  // ---- phase 4: per-row dual softmax + blend + log, one row per wave ----
  for (int r = wid; r < 16; r += NWAVES) {
    const float maskv = sMask[r];
    float gv[7], cv[7];
    float mg = -1e30f, mc = -1e30f;
    int cnt = 0;
    for (int c = lane; c < HIDDEN; c += 32) {
      float g  = sG[r * H_PAD + c];
      float cc = tanhf(sC[r * H_PAD + c]) + maskv;
      gv[cnt] = g; cv[cnt] = cc; ++cnt;
      mg = fmaxf(mg, g); mc = fmaxf(mc, cc);
    }
    mg = wmaxred(mg); mc = wmaxred(mc);
    float sg = 0.0f, sc = 0.0f;
    for (int i = 0; i < cnt; ++i) {
      float eg = __expf(gv[i] - mg);
      float ec = __expf(cv[i] - mc);
      gv[i] = eg; cv[i] = ec; sg += eg; sc += ec;
    }
    sg = wsumred(sg); sc = wsumred(sc);
    const float ig = 1.0f / sg, ic = 1.0f / sc;
    float* orow = out + (size_t)(mt * 16 + r) * HIDDEN;
    for (int i = 0; i < cnt; ++i) {
      int c = lane + 32 * i;
      orow[c] = __logf(0.5f * (gv[i] * ig + cv[i] * ic));  // ALPHA = 0.5 blend
    }
  }
}

// ---------------- launcher ----------------

extern "C" void kernel_launch(void* const* d_in, const int* in_sizes, int n_in,
                              void* d_out, int out_size, void* d_ws, size_t ws_size,
                              hipStream_t stream) {
  const float* ent  = (const float*)d_in[0];
  const float* relE = (const float*)d_in[1];
  const float* tim  = (const float*)d_in[2];
  const float* g_w  = (const float*)d_in[3];
  const float* g_b  = (const float*)d_in[4];
  const float* c_w  = (const float*)d_in[5];
  const float* c_b  = (const float*)d_in[6];
  const int*   hs   = (const int*)d_in[7];
  const int*   hr   = (const int*)d_in[8];
  // d_in[9] = hist_obj: unused (mask only needs (sub,rel) existence)
  const int*   sub  = (const int*)d_in[10];
  const int*   rel  = (const int*)d_in[11];

  char* ws = (char*)d_ws;                                  // all offsets 16B-aligned
  unsigned* pairFlag = (unsigned*)(ws + 0);                // 400000 B
  _Float16* gWt      = (_Float16*)(ws + 400000);           // 252928 B
  _Float16* cWt      = (_Float16*)(ws + 652928);           // 252928 B
  _Float16* entH     = (_Float16*)(ws + 905856);           // 400000 B
  _Float16* relH     = (_Float16*)(ws + 1305856);          //  40000 B
  _Float16* timH     = (_Float16*)(ws + 1345856);          //    400 B

  zero_u32<<<(NUM_ENTS * NUM_RELS + 255) / 256, 256, 0, stream>>>(
      pairFlag, NUM_ENTS * NUM_RELS);
  scatter_pairs<<<(NHIST + 255) / 256, 256, 0, stream>>>(hs, hr, pairFlag, NHIST);
  pack_wt<<<(H_PAD * K_PAD + 255) / 256, 256, 0, stream>>>(g_w, gWt);
  pack_wt<<<(H_PAD * K_PAD + 255) / 256, 256, 0, stream>>>(c_w, cWt);
  cvt_f16k<<<(NUM_ENTS * HIDDEN + 255) / 256, 256, 0, stream>>>(ent, entH, NUM_ENTS * HIDDEN);
  cvt_f16k<<<(NUM_RELS * HIDDEN + 255) / 256, 256, 0, stream>>>(relE, relH, NUM_RELS * HIDDEN);
  cvt_f16k<<<(HIDDEN + 255) / 256, 256, 0, stream>>>(tim, timH, HIDDEN);

  cygnet_main<<<NQ / 16, TPB, 0, stream>>>(entH, relH, timH, gWt, cWt, g_b, c_b,
                                           pairFlag, sub, rel, (float*)d_out);
}
